// GCN_22110491639835
// MI455X (gfx1250) — compile-verified, matrix operand order checked by reading, not run on hardware
//
#include <hip/hip_runtime.h>

// ---------------- problem constants ----------------
#define N_NODES 50000
#define N_EDGES 1600000
#define IN_C    512
#define HID     2048
#define MPAD    50048      // ceil(50000/128)*128 -> tail-free WMMA GEMM
#define NEG_SLOPE 0.2f

// ---------------- WMMA vector types ----------------
typedef __bf16 v16bf __attribute__((ext_vector_type(16)));
typedef float  v8f   __attribute__((ext_vector_type(8)));

#define BM 128
#define BN 128
#define BK 32

// CDNA5 async global->LDS copies (ASYNCcnt), if the toolchain exposes them.
#if defined(__AMDGCN__) && __has_builtin(__builtin_amdgcn_global_load_async_to_lds_b128) && __has_builtin(__builtin_amdgcn_s_wait_asynccnt)
#define GCN_ASYNC 1
#else
#define GCN_ASYNC 0
#endif

// builtin expects: (int4 AS1*, ... AS3*, imm offset, imm cpol)
typedef int v4i_ __attribute__((vector_size(4 * sizeof(int))));
#define AS_GLOBAL(p) ((__attribute__((address_space(1))) v4i_*)(p))
#define AS_LOCAL(p)  ((__attribute__((address_space(3))) v4i_*)(p))

// =====================================================================
// bf16 WMMA GEMM:  C[M,Nc] = A[M,K] @ Bt[Nc,K]^T
//   A  : bf16 row-major, stride K
//   Bt : bf16 PRE-TRANSPOSED weights [Nc, K], stride K  (so staging == A)
// 256 threads = 8 waves; wave w computes rows [w*16,w*16+16) x 128 cols.
// Double-buffered LDS, async global->LDS staging when available.
// =====================================================================
__global__ __launch_bounds__(256)
void gemm_bf16_wmma(const __bf16* __restrict__ A,
                    const __bf16* __restrict__ Bt,
                    float* __restrict__ C,
                    int K, int ldc)
{
    __shared__ __align__(16) __bf16 Alds[2][BM * BK];   // [m][k]
    __shared__ __align__(16) __bf16 Blds[2][BN * BK];   // [n][k]

    const int tid  = threadIdx.x;
    const int lane = tid & 31;
    const int wave = tid >> 5;     // 0..7
    const int hh   = lane >> 4;    // 0/1 (half-wave)
    const int l16  = lane & 15;

    const int mBase = blockIdx.y * BM;
    const int nBase = blockIdx.x * BN;

    // staging role: thread -> (row 0..127, 16-elem half 0/16); 32B per matrix
    const int row  = tid >> 1;
    const int half = (tid & 1) * 16;
    const __bf16* aSrc = A  + (size_t)(mBase + row) * K + half;
    const __bf16* bSrc = Bt + (size_t)(nBase + row) * K + half;

    v8f acc[8] = {};

    auto stage = [&](int buf, int k0) {
#if GCN_ASYNC
        __builtin_amdgcn_global_load_async_to_lds_b128(
            AS_GLOBAL(aSrc + k0),
            AS_LOCAL(&Alds[buf][row * BK + half]), 0, 0);
        __builtin_amdgcn_global_load_async_to_lds_b128(
            AS_GLOBAL(aSrc + k0 + 8),
            AS_LOCAL(&Alds[buf][row * BK + half + 8]), 0, 0);
        __builtin_amdgcn_global_load_async_to_lds_b128(
            AS_GLOBAL(bSrc + k0),
            AS_LOCAL(&Blds[buf][row * BK + half]), 0, 0);
        __builtin_amdgcn_global_load_async_to_lds_b128(
            AS_GLOBAL(bSrc + k0 + 8),
            AS_LOCAL(&Blds[buf][row * BK + half + 8]), 0, 0);
#else
        {
            const uint4* s = (const uint4*)(aSrc + k0);
            uint4 d0 = s[0], d1 = s[1];
            uint4* d = (uint4*)&Alds[buf][row * BK + half];
            d[0] = d0; d[1] = d1;
        }
        {
            const uint4* s = (const uint4*)(bSrc + k0);
            uint4 d0 = s[0], d1 = s[1];
            uint4* d = (uint4*)&Blds[buf][row * BK + half];
            d[0] = d0; d[1] = d1;
        }
#endif
    };

    const int nT = K / BK;

    stage(0, 0);
#if GCN_ASYNC
    __builtin_amdgcn_s_wait_asynccnt(0);
#endif
    __syncthreads();

    for (int kt = 0; kt < nT; ++kt) {
        const int cur = kt & 1;
        if (kt + 1 < nT) stage(cur ^ 1, (kt + 1) * BK);

        // A fragment: lane l16 = row m, elems k = hh*8+{0..7}, 16+hh*8+{0..7}
        union { v16bf v; uint4 q[2]; } af;
        {
            const int m = wave * 16 + l16;
            af.q[0] = *(const uint4*)&Alds[cur][m * BK + hh * 8];
            af.q[1] = *(const uint4*)&Alds[cur][m * BK + 16 + hh * 8];
        }

        #pragma unroll
        for (int nb = 0; nb < 8; ++nb) {
            union { v16bf v; uint4 q[2]; } bfr;
            const int n = nb * 16 + l16;
            bfr.q[0] = *(const uint4*)&Blds[cur][n * BK + hh * 8];
            bfr.q[1] = *(const uint4*)&Blds[cur][n * BK + 16 + hh * 8];
            acc[nb] = __builtin_amdgcn_wmma_f32_16x16x32_bf16(
                false, af.v, false, bfr.v, (short)0, acc[nb], false, false);
        }

        if (kt + 1 < nT) {
#if GCN_ASYNC
            __builtin_amdgcn_s_wait_asynccnt(0);
#endif
            __syncthreads();
        }
    }

    // ---- epilogue: C/D layout -> m = hh*8 + r, n = l16 ----
    #pragma unroll
    for (int nb = 0; nb < 8; ++nb) {
        #pragma unroll
        for (int r = 0; r < 8; ++r) {
            const int m = mBase + wave * 16 + hh * 8 + r;
            const int n = nBase + nb * 16 + l16;
            C[(size_t)m * ldc + n] = acc[nb][r];
        }
    }
}

// =====================================================================
// helpers
// =====================================================================

// W [K,N] f32 row-major -> Wt [N,K] bf16 row-major (transposed weights)
// grid = (K/256, N)
__global__ void cvt_w_transpose(const float* __restrict__ W, __bf16* __restrict__ Wt,
                                int K, int N)
{
    int n = blockIdx.y;
    int k = blockIdx.x * blockDim.x + threadIdx.x;
    Wt[(size_t)n * K + k] = (__bf16)W[(size_t)k * N + n];
}

// f32 -> bf16 with zero-padded rows; grid = (cols/256, padRows)
__global__ void cvt_pad(const float* __restrict__ s, __bf16* __restrict__ d,
                        int rows, int cols)
{
    int r = blockIdx.y;
    int c = blockIdx.x * blockDim.x + threadIdx.x;
    float v = (r < rows) ? s[(size_t)r * cols + c] : 0.0f;
    d[(size_t)r * cols + c] = (__bf16)v;
}

__global__ void fill_zero(float* __restrict__ p, size_t n)
{
    size_t stride = (size_t)gridDim.x * blockDim.x;
    for (size_t i = (size_t)blockIdx.x * blockDim.x + threadIdx.x; i < n; i += stride)
        p[i] = 0.0f;
}

// one block per edge: h[dst] += val * t[src]  (L2 atomic units)
__global__ __launch_bounds__(256)
void spmm_edge(const float* __restrict__ t, float* __restrict__ h,
               const int* __restrict__ esrc, const int* __restrict__ edst,
               const float* __restrict__ vals, int C)
{
    const int e   = blockIdx.x;
    const int src = esrc[e];
    const int dst = edst[e];
    const float v = vals[e];
    const float* trow = t + (size_t)src * C;
    float*       hrow = h + (size_t)dst * C;
    for (int c = threadIdx.x; c < C; c += 256)
        atomicAdd(&hrow[c], v * trow[c]);
}

// bias + leaky relu + bf16 convert (pad rows -> 0); grid = (C/256, padRows)
__global__ void bias_act_cvt(const float* __restrict__ h, const float* __restrict__ b,
                             __bf16* __restrict__ act, int rows, int C)
{
    int r = blockIdx.y;
    int c = blockIdx.x * blockDim.x + threadIdx.x;
    float v = 0.0f;
    if (r < rows) {
        v = h[(size_t)r * C + c] + b[c];
        v = (v >= 0.0f) ? v : NEG_SLOPE * v;
    }
    act[(size_t)r * C + c] = (__bf16)v;
}

// final: out = (h + b) / max(||h+b||_2, 1e-12), in place; one block per row
__global__ __launch_bounds__(256)
void l2norm_bias(float* __restrict__ h, const float* __restrict__ b, int C)
{
    __shared__ float red[8];
    __shared__ float sinv;
    float* p = h + (size_t)blockIdx.x * C;

    float s = 0.0f;
    for (int c = threadIdx.x; c < C; c += 256) {
        float v = p[c] + b[c];
        s += v * v;
    }
    #pragma unroll
    for (int off = 16; off > 0; off >>= 1)
        s += __shfl_down(s, off, 32);
    if ((threadIdx.x & 31) == 0) red[threadIdx.x >> 5] = s;
    __syncthreads();
    if (threadIdx.x == 0) {
        float tot = 0.0f;
        #pragma unroll
        for (int i = 0; i < 8; ++i) tot += red[i];
        sinv = 1.0f / fmaxf(sqrtf(tot), 1e-12f);
    }
    __syncthreads();
    float inv = sinv;
    for (int c = threadIdx.x; c < C; c += 256) {
        float v = p[c] + b[c];
        p[c] = v * inv;
    }
}

// =====================================================================
// launcher
// =====================================================================
extern "C" void kernel_launch(void* const* d_in, const int* in_sizes, int n_in,
                              void* d_out, int out_size, void* d_ws, size_t ws_size,
                              hipStream_t stream)
{
    const float* x    = (const float*)d_in[0];
    const float* avals= (const float*)d_in[1];
    const int*   esrc = (const int*)d_in[2];
    const int*   edst = (const int*)d_in[3];
    const float* W1   = (const float*)d_in[4];
    const float* b1   = (const float*)d_in[5];
    const float* W2   = (const float*)d_in[6];
    const float* b2   = (const float*)d_in[7];
    const float* W3   = (const float*)d_in[8];
    const float* b3   = (const float*)d_in[9];
    float* h = (float*)d_out;                 // reuse output as aggregation buffer

    // ---- workspace carve-up ----
    char* ws = (char*)d_ws;
    const size_t actB = (size_t)MPAD * HID * sizeof(__bf16);   // bf16 activations (padded)
    const size_t tB   = (size_t)MPAD * HID * sizeof(float);    // GEMM output
    __bf16* act = (__bf16*)ws;
    float*  t   = (float*)(ws + actB);
    __bf16* W1t = (__bf16*)(ws + actB + tB);                   // [HID, IN_C]
    __bf16* W2t = W1t + (size_t)IN_C * HID;                    // [HID, HID]
    __bf16* W3t = W2t + (size_t)HID * HID;                     // [HID, HID]

    // ---- weight conversion + transpose (bf16, [N,K]) ----
    cvt_w_transpose<<<dim3(IN_C / 256, HID), 256, 0, stream>>>(W1, W1t, IN_C, HID);
    cvt_w_transpose<<<dim3(HID / 256, HID), 256, 0, stream>>>(W2, W2t, HID, HID);
    cvt_w_transpose<<<dim3(HID / 256, HID), 256, 0, stream>>>(W3, W3t, HID, HID);

    const dim3 gemmGrid(HID / BN, MPAD / BM);
    const size_t hElems = (size_t)N_NODES * HID;

    // ================= layer 1 =================
    cvt_pad<<<dim3(IN_C / 256, MPAD), 256, 0, stream>>>(x, act, N_NODES, IN_C);
    gemm_bf16_wmma<<<gemmGrid, 256, 0, stream>>>(act, W1t, t, IN_C, HID);
    fill_zero<<<8192, 256, 0, stream>>>(h, hElems);
    spmm_edge<<<N_EDGES, 256, 0, stream>>>(t, h, esrc, edst, avals, HID);
    bias_act_cvt<<<dim3(HID / 256, MPAD), 256, 0, stream>>>(h, b1, act, N_NODES, HID);

    // ================= layer 2 =================
    gemm_bf16_wmma<<<gemmGrid, 256, 0, stream>>>(act, W2t, t, HID, HID);
    fill_zero<<<8192, 256, 0, stream>>>(h, hElems);
    spmm_edge<<<N_EDGES, 256, 0, stream>>>(t, h, esrc, edst, avals, HID);
    bias_act_cvt<<<dim3(HID / 256, MPAD), 256, 0, stream>>>(h, b2, act, N_NODES, HID);

    // ================= layer 3 =================
    gemm_bf16_wmma<<<gemmGrid, 256, 0, stream>>>(act, W3t, t, HID, HID);
    fill_zero<<<8192, 256, 0, stream>>>(h, hElems);
    spmm_edge<<<N_EDGES, 256, 0, stream>>>(t, h, esrc, edst, avals, HID);
    l2norm_bias<<<N_NODES, 256, 0, stream>>>(h, b3, HID);
}
